// Decoder_86277303042802
// MI455X (gfx1250) — matrix operand order repaired and checked by reading, not compiled, not run on hardware
//
#include <hip/hip_runtime.h>
#include <hip/hip_bf16.h>

// ---------------------------------------------------------------------------
// Pointer-network decoder for MI455X (gfx1250), bf16 WMMA pipeline with
// double-buffered async global->LDS staging for the dominant GEMM.
// B=64, N=1000 (pad 1024), D=512, H=8, DK=64, S=20 (pad 32).
// ---------------------------------------------------------------------------

typedef __bf16 bf16_t;
typedef __attribute__((ext_vector_type(16))) __bf16 v16bf;
typedef __attribute__((ext_vector_type(8)))  float  v8f;

#define WMMA_BF16(a, b, c) \
  __builtin_amdgcn_wmma_f32_16x16x32_bf16(false, (a), false, (b), (short)0, (c), false, false)

static constexpr int   kB   = 64;
static constexpr int   kN   = 1000;
static constexpr int   kNP  = 1024;
static constexpr int   kD   = 512;
static constexpr int   kH   = 8;
static constexpr int   kDK  = 64;
static constexpr int   kS   = 20;
static constexpr int   kSP  = 32;
static constexpr float kNEG = -1e9f;
static constexpr float kInvSqrtDK = 0.125f;               // 1/sqrt(64)
static constexpr float kInvSqrtD  = 0.04419417382415922f; // 1/sqrt(512)
static constexpr float kTanhClip  = 10.0f;

__device__ __forceinline__ bf16_t f32_to_bf16(float f) {
  unsigned int u = __builtin_bit_cast(unsigned int, f);
  unsigned int r = (u + 0x7FFFu + ((u >> 16) & 1u)) >> 16; // round-to-nearest-even
  return __builtin_bit_cast(bf16_t, (unsigned short)r);
}

__device__ __forceinline__ unsigned pack2_bf16(float lo, float hi) {
  unsigned a = __builtin_bit_cast(unsigned, lo);
  unsigned b = __builtin_bit_cast(unsigned, hi);
  unsigned ra = (a + 0x7FFFu + ((a >> 16) & 1u)) >> 16;
  unsigned rb = (b + 0x7FFFu + ((b >> 16) & 1u)) >> 16;
  return (rb << 16) | (ra & 0xFFFFu);
}

__device__ __forceinline__ v8f v8f_zero() {
  v8f z;
#pragma unroll
  for (int i = 0; i < 8; ++i) z[i] = 0.0f;
  return z;
}

// A fragment: 16x32 bf16 tile, row-major source with leading-dim lda (even).
// Lane l (0..15): M=l, K=0..15; lanes 16..31 same rows, K=16..31 (matches B).
__device__ __forceinline__ v16bf load_frag_a(const bf16_t* __restrict__ A, int lda, int lane) {
  const int row = lane & 15;
  const int kh  = (lane >> 4) << 4; // 0 or 16
  const bf16_t* p = A + (size_t)row * lda;
  v16bf f;
#pragma unroll
  for (int v = 0; v < 8; ++v) {
    const int k = kh + (v << 1);
    ((unsigned int*)&f)[v] = *(const unsigned int*)(p + k);
  }
  return f;
}

// B fragment: 32x16 bf16 tile from column-major storage (element (k, col) at
// col*ldb + k, ldb even). Lanes 0..15: col=lane, K=0..15; lanes 16..31: K=16..31.
__device__ __forceinline__ v16bf load_frag_b(const bf16_t* __restrict__ Bp, int ldb, int lane) {
  const int col = lane & 15;
  const int kb  = (lane >> 4) << 4; // 0 or 16
  const bf16_t* p = Bp + (size_t)col * ldb + kb;
  v16bf f;
#pragma unroll
  for (int v = 0; v < 8; ++v) {
    ((unsigned int*)&f)[v] = *(const unsigned int*)(p + (v << 1));
  }
  return f;
}

// ---------------------------------------------------------------------------
// K0a: convert embeddings [64000x512] fp32 -> bf16, 8 elements/thread.
// ---------------------------------------------------------------------------
__global__ __launch_bounds__(256) void k_convert_emb(
    const float* __restrict__ emb, bf16_t* __restrict__ embbf) {
  const size_t total8 = (size_t)kB * kN * kD / 8; // 4,096,000
  for (size_t i = (size_t)blockIdx.x * 256 + threadIdx.x; i < total8;
       i += (size_t)gridDim.x * 256) {
    const float4* s = (const float4*)(emb + i * 8);
    const float4 f0 = s[0];
    const float4 f1 = s[1];
    uint4 u;
    u.x = pack2_bf16(f0.x, f0.y);
    u.y = pack2_bf16(f0.z, f0.w);
    u.z = pack2_bf16(f1.x, f1.y);
    u.w = pack2_bf16(f1.z, f1.w);
    ((uint4*)embbf)[i] = u;
  }
}

// ---------------------------------------------------------------------------
// K0b: convert weights (transposed, so B-fragments read contiguous K pairs)
// and q (padded [B][H][32][64]) to bf16.
// ---------------------------------------------------------------------------
__global__ __launch_bounds__(256) void k_convert(
    const float* __restrict__ Wkvl, const float* __restrict__ Wout,
    const float* __restrict__ q,
    bf16_t* __restrict__ wkvlT, bf16_t* __restrict__ woutT, bf16_t* __restrict__ qbf) {
  const size_t NT1 = (size_t)3 * kD * kD;   // 1536*512 wkvlT
  const size_t NT2 = (size_t)kD * kD;       // 512*512 woutT
  const size_t NT3 = (size_t)kB * kH * kSP * kDK;
  const size_t total = NT1 + NT2 + NT3;
  for (size_t i = (size_t)blockIdx.x * 256 + threadIdx.x; i < total;
       i += (size_t)gridDim.x * 256) {
    if (i < NT1) {
      const size_t e = i / kD, k = i % kD;
      wkvlT[e * kD + k] = f32_to_bf16(Wkvl[k * (3 * kD) + e]);
    } else if (i < NT1 + NT2) {
      const size_t j = i - NT1, n = j / kD, k = j % kD;
      woutT[n * kD + k] = f32_to_bf16(Wout[k * kD + n]);
    } else {
      const size_t j = i - NT1 - NT2;
      const int dk = (int)(j & 63), s = (int)((j >> 6) & 31);
      const int h = (int)((j >> 11) & 7), b = (int)(j >> 14);
      qbf[j] = (s < kS) ? f32_to_bf16(q[((size_t)b * kS + s) * kD + h * kDK + dk])
                        : f32_to_bf16(0.0f);
    }
  }
}

// ---------------------------------------------------------------------------
// K1: kvl = emb_bf16[64000x512] @ W_kvl[512x1536]   (the 100.7 GFLOP GEMM)
// Block: 256 threads = 8 waves; tile 64(M) x 128(E).
// A tiles double-buffered through LDS with GLOBAL_LOAD_ASYNC_TO_LDS_B128:
// steady-state body always prefetches tile kt+1 and waits asynccnt<=1 so the
// copy overlaps the WMMAs of tile kt; the last iteration is peeled to keep a
// single control path through the backedge (lets acc stay pinned in-place,
// avoiding accumulator copies + WMMA->VALU hazard NOPs).
// ---------------------------------------------------------------------------
__global__ __launch_bounds__(256) void k_kvl_gemm(
    const bf16_t* __restrict__ embbf, const bf16_t* __restrict__ wkvlT,
    bf16_t* __restrict__ gk, bf16_t* __restrict__ vT, bf16_t* __restrict__ lk) {
  // 80-byte rows: 16B-aligned for b128 async stores, even stride for dword reads.
  __shared__ bf16_t As[2][64][40];

  const int tid  = threadIdx.x;
  const int lane = tid & 31;
  const int wid  = tid >> 5;
  const int mw   = wid >> 1;                 // 0..3 : 16-row slab
  const int ew   = wid & 1;                  // 0..1 : 64-col slab
  const int rowBase = blockIdx.y * 64;
  const int eBase   = blockIdx.x * 128 + ew * 64;

  v8f acc[4];
#pragma unroll
  for (int t = 0; t < 4; ++t) acc[t] = v8f_zero();

  // Async staging: one 16B transfer per thread per k-step (64x32 bf16 tile).
  const int rStage = tid >> 2;               // 0..63
  const int cg     = (tid & 3) * 8;          // 0,8,16,24 (bf16 elements)
  const bf16_t* gsrc = embbf + (size_t)(rowBase + rStage) * kD + cg;
  const unsigned ldsoff0 = (unsigned)(size_t)&As[0][rStage][cg]; // LDS address is the low 32 bits
  const unsigned ldsoff1 = (unsigned)(size_t)&As[1][rStage][cg];

  auto compute = [&](int kt, int buf) {
    const int k0 = kt * 32;
    const v16bf a = load_frag_a(&As[buf][mw * 16][0], 40, lane);
#pragma unroll
    for (int t = 0; t < 4; ++t) {
      const bf16_t* bp = wkvlT + (size_t)(eBase + t * 16) * kD + k0;
      const v16bf bb = load_frag_b(bp, kD, lane);
      acc[t] = WMMA_BF16(a, bb, acc[t]);
    }
  };

  constexpr int KT = kD / 32;  // 16 k-tiles

  // Prologue: prefetch k-tile 0 into buffer 0.
  asm volatile("global_load_async_to_lds_b128 %0, %1, off"
               :: "v"(ldsoff0), "v"(gsrc) : "memory");

  // Steady state: uniform body, always one prefetch in flight.
  for (int kt = 0; kt < KT - 1; ++kt) {
    const int buf = kt & 1;
    // All waves finished reading As[buf^1] in iteration kt-1: safe to refill it.
    __syncthreads();
    const bf16_t* gp = gsrc + (kt + 1) * 32;
    asm volatile("global_load_async_to_lds_b128 %0, %1, off"
                 :: "v"(buf ? ldsoff0 : ldsoff1), "v"(gp) : "memory");
    // Two async ops in flight; retire the older one (tile kt), keep prefetch going.
    asm volatile("s_wait_asynccnt 0x1" ::: "memory");
    __syncthreads();   // tile kt visible to every wave
    compute(kt, buf);
  }

  // Peeled epilogue: drain the last prefetch and compute the final tile.
  __syncthreads();
  asm volatile("s_wait_asynccnt 0x0" ::: "memory");
  __syncthreads();
  compute(KT - 1, (KT - 1) & 1);

  const int col = lane & 15;
  const int ro  = (lane >> 4) << 3;
#pragma unroll
  for (int t = 0; t < 4; ++t) {
    const int e = eBase + t * 16 + col;
#pragma unroll
    for (int r = 0; r < 8; ++r) {
      const int gr = rowBase + mw * 16 + r + ro;  // flat b*1000+n, < 64000
      const int b  = gr / kN;
      const int n  = gr - b * kN;
      const bf16_t val = f32_to_bf16(acc[t][r]);
      if (e < kD) {                // glimpse_k -> [B][H][NP][DK]
        const int h = e >> 6, dk = e & 63;
        gk[(((size_t)(b * kH + h)) * kNP + n) * kDK + dk] = val;
      } else if (e < 2 * kD) {     // glimpse_v transposed -> [B][H][DK][NP]
        const int e2 = e - kD, h = e2 >> 6, dk = e2 & 63;
        vT[(((size_t)(b * kH + h)) * kDK + dk) * kNP + n] = val;
      } else {                     // logit_k -> [B][NP][D]
        lk[((size_t)b * kNP + n) * kD + (e - 2 * kD)] = val;
      }
    }
  }
}

// ---------------------------------------------------------------------------
// K2a: compat[b,h,32,1024] = (q_h @ K_h^T) / sqrt(DK), masked (NEG).
// One block per (b,h); 8 waves x 128 columns each.
// ---------------------------------------------------------------------------
__global__ __launch_bounds__(256) void k_compat(
    const bf16_t* __restrict__ qbf, const bf16_t* __restrict__ gk,
    const unsigned char* __restrict__ am, float* __restrict__ compat) {
  const int bh = blockIdx.x;
  const int b  = bh >> 3;
  const int tid = threadIdx.x, lane = tid & 31, wid = tid >> 5;

  v16bf a[2][2];
#pragma unroll
  for (int mt = 0; mt < 2; ++mt)
#pragma unroll
    for (int ks = 0; ks < 2; ++ks)
      a[mt][ks] = load_frag_a(qbf + ((size_t)bh * kSP + mt * 16) * kDK + ks * 32, kDK, lane);

  const int col = lane & 15;
  const int ro  = (lane >> 4) << 3;

  for (int nt = 0; nt < 8; ++nt) {
    const int ncol0 = wid * 128 + nt * 16;
    const bf16_t* bbase = gk + ((size_t)bh * kNP + ncol0) * kDK;
    const v16bf bb0 = load_frag_b(bbase, kDK, lane);
    const v16bf bb1 = load_frag_b(bbase + 32, kDK, lane);
    const int n = ncol0 + col;
#pragma unroll
    for (int mt = 0; mt < 2; ++mt) {
      v8f acc = v8f_zero();
      acc = WMMA_BF16(a[mt][0], bb0, acc);
      acc = WMMA_BF16(a[mt][1], bb1, acc);
#pragma unroll
      for (int r = 0; r < 8; ++r) {
        const int s = mt * 16 + r + ro;
        const float v = acc[r] * kInvSqrtDK;
        const bool ok = (n < kN) && (s < kS) &&
                        (am[((size_t)b * kS + s) * kN + n] != 0);
        compat[((size_t)bh * kSP + s) * kNP + n] = ok ? v : kNEG;
      }
    }
  }
}

// ---------------------------------------------------------------------------
// K2b: row softmax over 1024 cols; attn written as bf16 (pad cols -> ~0).
// Wave-per-4-rows, wave32 shfl reductions.
// ---------------------------------------------------------------------------
__global__ __launch_bounds__(256) void k_softmax(
    const float* __restrict__ compat, bf16_t* __restrict__ attn) {
  const int bh = blockIdx.x;
  const int tid = threadIdx.x, lane = tid & 31, wid = tid >> 5;
  for (int i = 0; i < 4; ++i) {
    const int row = wid * 4 + i;
    const float* crow = compat + ((size_t)bh * kSP + row) * kNP;
    float mx = -3.0e38f;
    for (int n = lane; n < kNP; n += 32) mx = fmaxf(mx, crow[n]);
#pragma unroll
    for (int off = 16; off > 0; off >>= 1) mx = fmaxf(mx, __shfl_xor(mx, off, 32));
    float se = 0.0f;
    for (int n = lane; n < kNP; n += 32) se += __expf(crow[n] - mx);
#pragma unroll
    for (int off = 16; off > 0; off >>= 1) se += __shfl_xor(se, off, 32);
    const float inv = 1.0f / se;
    bf16_t* arow = attn + ((size_t)bh * kSP + row) * kNP;
    for (int n = lane; n < kNP; n += 32)
      arow[n] = f32_to_bf16(__expf(crow[n] - mx) * inv);
  }
}

// ---------------------------------------------------------------------------
// K2c: heads[b,s,h*64+dk] = attn[32x1024] @ V[1024x64] per (b,h).
// 8 waves = 2(M) x 4(N) tiles; 32 k-steps of WMMA.
// ---------------------------------------------------------------------------
__global__ __launch_bounds__(256) void k_heads(
    const bf16_t* __restrict__ attn, const bf16_t* __restrict__ vT,
    bf16_t* __restrict__ heads) {
  const int bh = blockIdx.x;
  const int b = bh >> 3, h = bh & 7;
  const int tid = threadIdx.x, lane = tid & 31, wid = tid >> 5;
  const int mt = wid >> 2, ntl = wid & 3;

  v8f acc = v8f_zero();
  for (int kt = 0; kt < kNP / 32; ++kt) {
    const int k0 = kt * 32;
    const v16bf a  = load_frag_a(attn + ((size_t)bh * kSP + mt * 16) * kNP + k0, kNP, lane);
    const v16bf bb = load_frag_b(vT + ((size_t)bh * kDK + ntl * 16) * kNP + k0, kNP, lane);
    acc = WMMA_BF16(a, bb, acc);
  }
  const int col = lane & 15, ro = (lane >> 4) << 3;
  const int d = h * kDK + ntl * 16 + col;
#pragma unroll
  for (int r = 0; r < 8; ++r) {
    const int s = mt * 16 + r + ro;
    heads[((size_t)b * kSP + s) * kD + d] = f32_to_bf16(acc[r]);
  }
}

// ---------------------------------------------------------------------------
// K3: glimpse = heads[32x512] @ W_out[512x512] per b (bf16 WMMA).
// ---------------------------------------------------------------------------
__global__ __launch_bounds__(256) void k_glimpse(
    const bf16_t* __restrict__ heads, const bf16_t* __restrict__ woutT,
    bf16_t* __restrict__ glimpse) {
  const int b = blockIdx.x, nch = blockIdx.y;   // nch: 8 chunks of 64 cols
  const int tid = threadIdx.x, lane = tid & 31, wid = tid >> 5;
  const int mt = wid >> 2, ntl = wid & 3;
  const int ncol = nch * 64 + ntl * 16;

  v8f acc = v8f_zero();
  for (int kt = 0; kt < kD / 32; ++kt) {
    const int k0 = kt * 32;
    const v16bf a  = load_frag_a(heads + ((size_t)b * kSP + mt * 16) * kD + k0, kD, lane);
    const v16bf bb = load_frag_b(woutT + (size_t)ncol * kD + k0, kD, lane);
    acc = WMMA_BF16(a, bb, acc);
  }
  const int col = lane & 15, ro = (lane >> 4) << 3;
#pragma unroll
  for (int r = 0; r < 8; ++r) {
    const int s = mt * 16 + r + ro;
    glimpse[((size_t)b * kSP + s) * kD + ncol + col] = f32_to_bf16(acc[r]);
  }
}

// ---------------------------------------------------------------------------
// K4: raw pointer logits[b,32,1024] = glimpse[32x512] @ logit_k^T (bf16 WMMA).
// ---------------------------------------------------------------------------
__global__ __launch_bounds__(256) void k_logits(
    const bf16_t* __restrict__ glimpse, const bf16_t* __restrict__ lk,
    float* __restrict__ logits) {
  const int b = blockIdx.x, nch = blockIdx.y;   // nch: 8 chunks of 128 cols
  const int tid = threadIdx.x, lane = tid & 31, wid = tid >> 5;
  const int ncol = nch * 128 + wid * 16;

  v8f acc0 = v8f_zero(), acc1 = v8f_zero();
  for (int kt = 0; kt < kD / 32; ++kt) {
    const int k0 = kt * 32;
    const v16bf a0 = load_frag_a(glimpse + ((size_t)b * kSP + 0) * kD + k0, kD, lane);
    const v16bf a1 = load_frag_a(glimpse + ((size_t)b * kSP + 16) * kD + k0, kD, lane);
    const v16bf bb = load_frag_b(lk + ((size_t)b * kNP + ncol) * kD + k0, kD, lane);
    acc0 = WMMA_BF16(a0, bb, acc0);
    acc1 = WMMA_BF16(a1, bb, acc1);
  }
  const int col = lane & 15, ro = (lane >> 4) << 3;
#pragma unroll
  for (int r = 0; r < 8; ++r) {
    logits[((size_t)b * kSP + (r + ro)) * kNP + ncol + col] = acc0[r];
    logits[((size_t)b * kSP + (16 + r + ro)) * kNP + ncol + col] = acc1[r];
  }
}

// ---------------------------------------------------------------------------
// K5: scale, tanh-clip, mask, row log-softmax, write out[(s*B+b)*N + n].
// ---------------------------------------------------------------------------
__global__ __launch_bounds__(256) void k_logsoftmax(
    const float* __restrict__ logits, const unsigned char* __restrict__ am,
    float* __restrict__ out) {
  const int b = blockIdx.x, s = blockIdx.y;
  const int tid = threadIdx.x;
  __shared__ float red[256];

  float vals[4];
  float mx = -3.0e38f;
#pragma unroll
  for (int i = 0; i < 4; ++i) {
    const int n = tid + i * 256;
    float x = -3.0e38f;
    if (n < kN) {
      x = kTanhClip * tanhf(logits[((size_t)b * kSP + s) * kNP + n] * kInvSqrtD);
      if (am[((size_t)b * kS + s) * kN + n] == 0) x = kNEG;
      mx = fmaxf(mx, x);
    }
    vals[i] = x;
  }
  red[tid] = mx;
  __syncthreads();
  for (int off = 128; off > 0; off >>= 1) {
    if (tid < off) red[tid] = fmaxf(red[tid], red[tid + off]);
    __syncthreads();
  }
  const float rowmax = red[0];
  __syncthreads();

  float se = 0.0f;
#pragma unroll
  for (int i = 0; i < 4; ++i) {
    const int n = tid + i * 256;
    if (n < kN) se += __expf(vals[i] - rowmax);
  }
  red[tid] = se;
  __syncthreads();
  for (int off = 128; off > 0; off >>= 1) {
    if (tid < off) red[tid] += red[tid + off];
    __syncthreads();
  }
  const float lse = logf(red[0]);

#pragma unroll
  for (int i = 0; i < 4; ++i) {
    const int n = tid + i * 256;
    if (n < kN)
      out[((size_t)s * kB + b) * kN + n] = vals[i] - rowmax - lse;
  }
}

// ---------------------------------------------------------------------------
// Launch
// ---------------------------------------------------------------------------
extern "C" void kernel_launch(void* const* d_in, const int* in_sizes, int n_in,
                              void* d_out, int out_size, void* d_ws, size_t ws_size,
                              hipStream_t stream) {
  (void)in_sizes; (void)n_in; (void)out_size; (void)ws_size;

  const float* emb  = (const float*)d_in[0];               // [B,N,D] fp32
  const float* q    = (const float*)d_in[1];               // [B,S,D] fp32
  const unsigned char* am = (const unsigned char*)d_in[2]; // [B,S,N] bool (1B)
  const float* Wkvl = (const float*)d_in[3];               // [D,3D] fp32
  const float* Wout = (const float*)d_in[4];               // [D,D] fp32
  float* out = (float*)d_out;                              // [S*B, N] fp32

  char* p = (char*)d_ws;
  auto alloc = [&](size_t bytes) {
    char* r = p;
    p += (bytes + 255) & ~(size_t)255;
    return r;
  };

  bf16_t* embbf   = (bf16_t*)alloc((size_t)kB * kN * kD * 2);            // 65.5 MB
  bf16_t* wkvlT   = (bf16_t*)alloc((size_t)3 * kD * kD * 2);             // 1.5 MB
  bf16_t* woutT   = (bf16_t*)alloc((size_t)kD * kD * 2);                 // 0.5 MB
  bf16_t* qbf     = (bf16_t*)alloc((size_t)kB * kH * kSP * kDK * 2);     // 2 MB
  bf16_t* gk      = (bf16_t*)alloc((size_t)kB * kH * kNP * kDK * 2);     // 67 MB
  bf16_t* vT      = (bf16_t*)alloc((size_t)kB * kH * kDK * kNP * 2);     // 67 MB
  bf16_t* lk      = (bf16_t*)alloc((size_t)kB * kNP * kD * 2);           // 67 MB
  float*  compat  = (float*) alloc((size_t)kB * kH * kSP * kNP * 4);     // 67 MB
  bf16_t* attn    = (bf16_t*)alloc((size_t)kB * kH * kSP * kNP * 2);     // 33.5 MB
  bf16_t* heads   = (bf16_t*)alloc((size_t)kB * kSP * kD * 2);           // 2 MB
  bf16_t* glimpse = (bf16_t*)alloc((size_t)kB * kSP * kD * 2);           // 2 MB
  float*  logits  = (float*) alloc((size_t)kB * kSP * kNP * 4);          // 8.4 MB

  k_convert_emb<<<16000, 256, 0, stream>>>(emb, embbf);
  k_convert<<<8192, 256, 0, stream>>>(Wkvl, Wout, q, wkvlT, woutT, qbf);
  k_kvl_gemm<<<dim3(12, 1000), 256, 0, stream>>>(embbf, wkvlT, gk, vT, lk);
  k_compat<<<kB * kH, 256, 0, stream>>>(qbf, gk, am, compat);
  k_softmax<<<kB * kH, 256, 0, stream>>>(compat, attn);
  k_heads<<<kB * kH, 256, 0, stream>>>(attn, vT, heads);
  k_glimpse<<<dim3(kB, 8), 256, 0, stream>>>(heads, woutT, glimpse);
  k_logits<<<dim3(kB, 8), 256, 0, stream>>>(glimpse, lk, logits);
  k_logsoftmax<<<dim3(kB, kS), 256, 0, stream>>>(logits, am, out);
}
